// SimpleHAN_82454782148688
// MI455X (gfx1250) — matrix operand order repaired and checked by reading, not compiled, not run on hardware
//
#include <hip/hip_runtime.h>
#include <hip/hip_bf16.h>
#include <math.h>

// ---------------- problem constants (match reference) ----------------
#define NN      50000      // nodes
#define INCH    128
#define HIDC    256
#define HEADS   8
#define HDIM    32         // HID / HEADS
#define NE      800000     // edges per metapath
#define ET      (NE + NN)  // edges + self loops
#define NEG_SLOPE 0.2f

typedef __attribute__((ext_vector_type(16))) __bf16 v16bf;
typedef __attribute__((ext_vector_type(8)))  __bf16 v8bf;
typedef __attribute__((ext_vector_type(8)))  float  v8f;

// ---- monotone float<->uint mapping for atomic max on signed floats ----
__device__ __forceinline__ unsigned fkey(float f) {
    unsigned u = __float_as_uint(f);
    return (u & 0x80000000u) ? ~u : (u | 0x80000000u);
}
__device__ __forceinline__ float funkey(unsigned u) {
    unsigned v = (u & 0x80000000u) ? (u & 0x7fffffffu) : ~u;
    return __uint_as_float(v);
}

// ---------------------------------------------------------------------
// Pack weight B[K,Nc] (f32, row-major) -> Bt[Nc,K] (bf16, transposed).
// Weights are <=256KB, fully L2 resident; done once per GEMM.
// ---------------------------------------------------------------------
__global__ void han_pack_bt(const float* __restrict__ B, __bf16* __restrict__ Bt,
                            int K, int Nc)
{
    int idx = blockIdx.x * blockDim.x + threadIdx.x;
    if (idx >= K * Nc) return;
    int n = idx / K, k = idx - n * K;
    Bt[idx] = (__bf16)B[(size_t)k * Nc + n];
}

// =====================================================================
// bf16 WMMA GEMM:  C[M,Nc] = act(A[M,K] @ B[K,Nc] + bias)
// B is pre-packed transposed bf16 (Bt[Nc][K]) so each lane's B fragment
// is 2 contiguous b128 loads.
// block = 256 threads = 8 waves, waves tiled 4(M) x 2(N).
// Each wave computes a 16x64 strip of C (four 16x16 WMMA tiles sharing
// one A fragment) -> block tile = 64(M) x 128(N).
// K multiple of 32, Nc multiple of 128.
// =====================================================================
template <bool BIAS, bool RELU>
__global__ void han_gemm(const float* __restrict__ A,
                         const __bf16* __restrict__ Bt,
                         const float* __restrict__ bias,
                         float* __restrict__ C,
                         int M, int K, int Nc)
{
    const int lane = threadIdx.x & 31;
    const int wave = threadIdx.x >> 5;
    const int half = lane >> 4;      // 0: lanes 0-15, 1: lanes 16-31
    const int l16  = lane & 15;

    const int row0 = blockIdx.x * 64  + (wave & 3) * 16;   // C tile row base
    const int colb = blockIdx.y * 128 + (wave >> 2) * 64;  // C strip col base

    const int arow_g = row0 + l16;                        // A row this lane feeds
    const int arow_c = arow_g < M ? arow_g : (M - 1);     // clamp for tail tiles
    const float* __restrict__ arow = A + (size_t)arow_c * K;

    int ncol[4];
    const __bf16* __restrict__ bt[4];
#pragma unroll
    for (int t = 0; t < 4; ++t) {
        ncol[t] = colb + t * 16 + l16;                    // B/C column (tile t)
        // lane's fragment: Bt[ncol][k0 + half*16 .. +15] (contiguous bf16)
        bt[t] = Bt + (size_t)ncol[t] * K + half * 16;
    }

    v8f acc[4] = {};
    for (int k0 = 0; k0 < K; k0 += 32) {
        // ---- A fragment: contiguous K = half*16 + s (matches B fragment) ----
        v16bf a;
#pragma unroll
        for (int i = 0; i < 8; ++i) {
            a[i]     = (__bf16)arow[k0 + half * 16 + i];       // K = half*16 + i
            a[8 + i] = (__bf16)arow[k0 + half * 16 + 8 + i];   // K = half*16 + 8 + i
        }
        // ---- 4 B fragments + 4 WMMA (independent accumulators) ----
#pragma unroll
        for (int t = 0; t < 4; ++t) {
            const v8bf* bp = (const v8bf*)(bt[t] + k0);
            v8bf lo = bp[0], hi = bp[1];
            v16bf b = __builtin_shufflevector(lo, hi,
                0, 1, 2, 3, 4, 5, 6, 7, 8, 9, 10, 11, 12, 13, 14, 15);
            acc[t] = __builtin_amdgcn_wmma_f32_16x16x32_bf16(
                false, a, false, b, (short)0, acc[t], false, false);
        }
    }

    float bv[4] = {};
    if (BIAS) {
#pragma unroll
        for (int t = 0; t < 4; ++t) bv[t] = bias[ncol[t]];
    }

    // ---- store: C VGPR r holds M = r + 8*half (ISA 7.12.2) ----
#pragma unroll
    for (int r = 0; r < 8; ++r) {
        int row = row0 + half * 8 + r;
        if (row < M) {
            float* cr = C + (size_t)row * Nc;
#pragma unroll
            for (int t = 0; t < 4; ++t) {
                float v = acc[t][r];
                if (BIAS) v += bv[t];
                if (RELU) v = fmaxf(v, 0.f);
                cr[ncol[t]] = v;
            }
        }
    }
}

// ---------------------------------------------------------------------
// a_src[n,h] = <hW[n,h,:], att_src[h,:]> ; same for a_dst
// ---------------------------------------------------------------------
__global__ void han_attn_coef(const float* __restrict__ hW,
                              const float* __restrict__ aw_src,
                              const float* __restrict__ aw_dst,
                              float* __restrict__ asrc,
                              float* __restrict__ adst)
{
    int idx = blockIdx.x * blockDim.x + threadIdx.x;
    if (idx >= NN * HEADS) return;
    int n = idx >> 3, h = idx & 7;
    const float* hp = hW + (size_t)n * HIDC + h * HDIM;
    const float* ws = aw_src + h * HDIM;
    const float* wd = aw_dst + h * HDIM;
    float s = 0.f, d = 0.f;
#pragma unroll
    for (int i = 0; i < HDIM; ++i) {
        float v = hp[i];
        s += v * ws[i];
        d += v * wd[i];
    }
    asrc[idx] = s;
    adst[idx] = d;
}

// zero acc[N*HID], segmax(uint)[N*HEADS], segsum[N*HEADS]
__global__ void han_init(float* __restrict__ acc, unsigned* __restrict__ segmax,
                         float* __restrict__ segsum)
{
    int idx = blockIdx.x * blockDim.x + threadIdx.x;
    if (idx < NN * HIDC) acc[idx] = 0.f;
    if (idx < NN * HEADS) { segmax[idx] = 0u; segsum[idx] = 0.f; }
}

// pass 1: e = leaky_relu(a_src[src]+a_dst[dst]); segment max via uint atomicMax
__global__ void han_edge_max(const int* __restrict__ ei,
                             const float* __restrict__ asrc,
                             const float* __restrict__ adst,
                             float* __restrict__ ebuf,
                             unsigned* __restrict__ segmax)
{
    int idx = blockIdx.x * blockDim.x + threadIdx.x;
    if (idx >= ET * HEADS) return;
    int e = idx >> 3, h = idx & 7;
    int s, d;
    if (e < NE) { s = ei[e]; d = ei[NE + e]; }
    else        { s = d = e - NE; }
    float v = asrc[s * HEADS + h] + adst[d * HEADS + h];
    v = v > 0.f ? v : NEG_SLOPE * v;
    ebuf[idx] = v;
    atomicMax(&segmax[d * HEADS + h], fkey(v));
}

// pass 2: p = exp(e - max[dst]); segment sum
__global__ void han_edge_exp(const int* __restrict__ ei,
                             float* __restrict__ ebuf,
                             const unsigned* __restrict__ segmax,
                             float* __restrict__ segsum)
{
    int idx = blockIdx.x * blockDim.x + threadIdx.x;
    if (idx >= ET * HEADS) return;
    int e = idx >> 3, h = idx & 7;
    int d = (e < NE) ? ei[NE + e] : (e - NE);
    float m = funkey(segmax[d * HEADS + h]);
    float p = __expf(ebuf[idx] - m);
    ebuf[idx] = p;
    atomicAdd(&segsum[d * HEADS + h], p);
}

// pass 3: acc[dst,:] += hW[src,:] * alpha   (one 256-thread block per edge)
__global__ void han_edge_scatter(const int* __restrict__ ei,
                                 const float* __restrict__ ebuf,
                                 const float* __restrict__ segsum,
                                 const float* __restrict__ hW,
                                 float* __restrict__ acc)
{
    int e = blockIdx.x;
    int t = threadIdx.x;          // channel 0..255
    int h = t >> 5;               // head
    int s, d;
    if (e < NE) { s = ei[e]; d = ei[NE + e]; }
    else        { s = d = e - NE; }
    float alpha = ebuf[e * HEADS + h] / (segsum[d * HEADS + h] + 1e-16f);
    atomicAdd(&acc[(size_t)d * HIDC + t], hW[(size_t)s * HIDC + t] * alpha);
}

// out = elu(acc + bias)  (in place)
__global__ void han_bias_elu(float* __restrict__ acc, const float* __restrict__ bias)
{
    int idx = blockIdx.x * blockDim.x + threadIdx.x;
    if (idx >= NN * HIDC) return;
    float v = acc[idx] + bias[idx & (HIDC - 1)];
    acc[idx] = v > 0.f ? v : expm1f(v);
}

// z = softmax(sem_att)[0]*tut + [1]*tdt
__global__ void han_combine(const float* __restrict__ tut, const float* __restrict__ tdt,
                            const float* __restrict__ sem, float* __restrict__ z)
{
    int idx = blockIdx.x * blockDim.x + threadIdx.x;
    if (idx >= NN * HIDC) return;
    float s0 = sem[0], s1 = sem[1];
    float m  = fmaxf(s0, s1);
    float e0 = __expf(s0 - m), e1 = __expf(s1 - m);
    float inv = 1.f / (e0 + e1);
    z[idx] = (e0 * inv) * tut[idx] + (e1 * inv) * tdt[idx];
}

// logits[n,c] = hc[n,:] @ W2[:,c] + b2[c]   (K=128, Nc=2 -> plain VALU)
__global__ void han_logits(const float* __restrict__ hc, const float* __restrict__ W2,
                           const float* __restrict__ b2, float* __restrict__ out)
{
    int idx = blockIdx.x * blockDim.x + threadIdx.x;
    if (idx >= NN * 2) return;
    int n = idx >> 1, c = idx & 1;
    const float* hp = hc + (size_t)n * 128;
    float s = b2[c];
#pragma unroll 8
    for (int k = 0; k < 128; ++k) s += hp[k] * W2[k * 2 + c];
    out[idx] = s;
}

// =====================================================================
extern "C" void kernel_launch(void* const* d_in, const int* in_sizes, int n_in,
                              void* d_out, int out_size, void* d_ws, size_t ws_size,
                              hipStream_t stream)
{
    (void)in_sizes; (void)n_in; (void)out_size; (void)ws_size;

    const float* x      = (const float*)d_in[0];
    const int*   ei[2]  = { (const int*)d_in[1], (const int*)d_in[2] };   // tut, tdt
    const float* projW  = (const float*)d_in[3];
    const float* projB  = (const float*)d_in[4];
    const float* Wb[2]    = { (const float*)d_in[5],  (const float*)d_in[9]  };
    const float* asrcW[2] = { (const float*)d_in[6],  (const float*)d_in[10] };
    const float* adstW[2] = { (const float*)d_in[7],  (const float*)d_in[11] };
    const float* biasB[2] = { (const float*)d_in[8],  (const float*)d_in[12] };
    const float* sem    = (const float*)d_in[13];
    const float* clsW1  = (const float*)d_in[14];
    const float* clsB1  = (const float*)d_in[15];
    const float* clsW2  = (const float*)d_in[16];
    const float* clsB2  = (const float*)d_in[17];
    float* out = (float*)d_out;

    // ---------------- workspace layout (floats) ----------------
    const size_t NB = (size_t)NN * HIDC;     // 12.8M floats
    float* ws   = (float*)d_ws;
    float* h    = ws;                        // projected features
    float* bufA = ws + NB;                   // layer-0 output per branch
    float* hW   = ws + 2 * NB;               // h @ W  (also reused for classifier hidden)
    float* tut  = ws + 3 * NB;
    float* tdt  = ws + 4 * NB;
    float* asrc = ws + 5 * NB;
    float* adst = asrc + (size_t)NN * HEADS;
    unsigned* segmax = (unsigned*)(adst + (size_t)NN * HEADS);
    float* segsum    = (float*)(segmax + (size_t)NN * HEADS);
    float* ebuf      = segsum + (size_t)NN * HEADS;          // [ET*HEADS]
    __bf16* Bt       = (__bf16*)(ebuf + (size_t)ET * HEADS); // packed weight, <=128KB
    float* branch_out[2] = { tut, tdt };

    const int T = 256;
    const int gNH  = (NN * HIDC + T - 1) / T;
    const int gNh  = (NN * HEADS + T - 1) / T;
    const int gEh  = (ET * HEADS + T - 1) / T;
    const int gPK  = (HIDC * HIDC + T - 1) / T;   // covers largest weight

    // 1) projection: h = x @ projW + projB   [50000,128]x[128,256]
    {
        han_pack_bt<<<gPK, T, 0, stream>>>(projW, Bt, INCH, HIDC);
        dim3 grid((NN + 63) / 64, HIDC / 128);
        han_gemm<true, false><<<grid, T, 0, stream>>>(x, Bt, projB, h, NN, INCH, HIDC);
    }

    // 2) two metapath branches, 2 GAT layers each
    for (int b = 0; b < 2; ++b) {
        const float* hin = h;
        for (int l = 0; l < 2; ++l) {
            float* acc = (l == 0) ? bufA : branch_out[b];
            // hW = hin @ W[l]
            han_pack_bt<<<gPK, T, 0, stream>>>(Wb[b] + (size_t)l * HIDC * HIDC, Bt,
                                               HIDC, HIDC);
            dim3 grid((NN + 63) / 64, HIDC / 128);
            han_gemm<false, false><<<grid, T, 0, stream>>>(
                hin, Bt, nullptr, hW, NN, HIDC, HIDC);
            // attention coefficients
            han_attn_coef<<<gNh, T, 0, stream>>>(
                hW, asrcW[b] + l * HEADS * HDIM, adstW[b] + l * HEADS * HDIM, asrc, adst);
            // zero accumulators
            han_init<<<gNH, T, 0, stream>>>(acc, segmax, segsum);
            // segment softmax + scatter
            han_edge_max<<<gEh, T, 0, stream>>>(ei[b], asrc, adst, ebuf, segmax);
            han_edge_exp<<<gEh, T, 0, stream>>>(ei[b], ebuf, segmax, segsum);
            han_edge_scatter<<<ET, T, 0, stream>>>(ei[b], ebuf, segsum, hW, acc);
            // bias + elu (in place)
            han_bias_elu<<<gNH, T, 0, stream>>>(acc, biasB[b] + l * HIDC);
            hin = acc;
        }
    }

    // 3) semantic attention combine -> z (reuse bufA)
    han_combine<<<gNH, T, 0, stream>>>(tut, tdt, sem, bufA);

    // 4) classifier hidden: hc = relu(z @ clsW1 + clsB1)  [50000,256]x[256,128] -> hW
    {
        han_pack_bt<<<gPK, T, 0, stream>>>(clsW1, Bt, HIDC, 128);
        dim3 grid((NN + 63) / 64, 1);
        han_gemm<true, true><<<grid, T, 0, stream>>>(bufA, Bt, clsB1, hW, NN, HIDC, 128);
    }

    // 5) logits
    {
        int g = (NN * 2 + T - 1) / T;
        han_logits<<<g, T, 0, stream>>>(hW, clsW2, clsB2, out);
    }
}
